// MHSA_4698694222637
// MI455X (gfx1250) — matrix operand order repaired and checked by reading, not compiled, not run on hardware
//
#include <hip/hip_runtime.h>

typedef __attribute__((ext_vector_type(16))) _Float16 v16h;
typedef __attribute__((ext_vector_type(8)))  float    v8f;
typedef __attribute__((ext_vector_type(4)))  int      v4i;
typedef unsigned short u16_t;
typedef unsigned int   u32_t;

#define HEADS 8
#define DHEAD 64
#define CCH   512
#define NPIX  1024
#define BATCH 32

// Async global->LDS path (gfx1250): guarded so toolchain differences (and the
// host pass) still compile via the uint4 fallback.
#if defined(__gfx1250__) && __has_builtin(__builtin_amdgcn_global_load_async_to_lds_b128) && __has_builtin(__builtin_amdgcn_s_wait_asynccnt)
#define USE_ASYNC_LDS 1
#else
#define USE_ASYNC_LDS 0
#endif

#if USE_ASYNC_LDS
typedef __attribute__((address_space(1))) v4i* v4i_gptr;   // global int4*
typedef __attribute__((address_space(3))) v4i* v4i_lptr;   // LDS int4*
#endif

__device__ __forceinline__ u16_t f2h_bits(float f) {
  _Float16 h = (_Float16)f;
  return __builtin_bit_cast(u16_t, h);
}

__device__ __forceinline__ v8f wmma_f16(v16h a, v16h b, v8f c) {
  // (neg_a, A, neg_b, B, c_mod, C, reuse_a, reuse_b)
  return __builtin_amdgcn_wmma_f32_16x16x32_f16(false, a, false, b, (short)0, c,
                                                false, false);
}

// 16-byte global -> LDS copy: async DMA when available (ASYNCcnt tracked).
__device__ __forceinline__ void copy16(void* lds_dst, const void* gsrc) {
#if USE_ASYNC_LDS
  void* g = const_cast<void*>(gsrc);
  __builtin_amdgcn_global_load_async_to_lds_b128((v4i_gptr)g, (v4i_lptr)lds_dst,
                                                 0, 0);
#else
  *(uint4*)lds_dst = *(const uint4*)gsrc;
#endif
}

__device__ __forceinline__ void wait_async_copies() {
#if USE_ASYNC_LDS
  __builtin_amdgcn_s_wait_asynccnt(0);
#endif
}

union FragU { v16h v; u32_t u[8]; };

// A fragment (16Mx32K f16). LDS tile row-major [row][k], k-pairs contiguous.
// lane L -> row = L&15; hl = L>>4; VGPR i holds K = (i&3)*2 + (i>>2)*16 + hl*8.
__device__ __forceinline__ v16h load_a_frag(const u16_t* base, int row0, int k0,
                                            int stride, int lane) {
  FragU f;
  const int row = row0 + (lane & 15);
  const int hl  = lane >> 4;
  const u16_t* p = base + row * stride + k0 + (hl << 3);
#pragma unroll
  for (int i = 0; i < 8; ++i) {
    const int kk = ((i & 3) << 1) + ((i >> 2) << 4);
    f.u[i] = *(const u32_t*)(p + kk);
  }
  return f.v;
}

// B fragment (32Kx16N f16). LDS tile transposed [col][k].
// lane L -> col = L&15; hl = L>>4; VGPR i holds K = (i&3)*2 + (i>>2)*16 + hl*8.
__device__ __forceinline__ v16h load_b_frag(const u16_t* base, int col0, int k0,
                                            int stride, int lane) {
  FragU f;
  const int col = col0 + (lane & 15);
  const int hl  = lane >> 4;
  const u16_t* p = base + col * stride + k0 + (hl << 3);
#pragma unroll
  for (int i = 0; i < 8; ++i) f.u[i] = *(const u32_t*)(p + ((i & 3) << 1) + ((i >> 2) << 4));
  return f.v;
}

// ---------------------------------------------------------------------------
// Kernel 1: positional table, stored TRANSPOSED: Pt[h][n][d] (f16)
// pos[h,d,n] = rel_h[h,d,n%32] + rel_w[h,d,n/32]
// ---------------------------------------------------------------------------
__global__ __launch_bounds__(256) void pos_build(const float* __restrict__ rel_h,
                                                 const float* __restrict__ rel_w,
                                                 u16_t* __restrict__ Pt) {
  const int i = blockIdx.x * 256 + threadIdx.x;   // HEADS*NPIX*DHEAD
  const int d = i & 63;
  const int n = (i >> 6) & (NPIX - 1);
  const int h = i >> 16;
  const float v = rel_h[(h * DHEAD + d) * 32 + (n & 31)] +
                  rel_w[(h * DHEAD + d) * 32 + (n >> 5)];
  Pt[i] = f2h_bits(v);
}

// ---------------------------------------------------------------------------
// Kernel 2: QKV projection GEMM. Y[o,n] = sum_c W[o,c] X[b,c,n] + bias[o].
// Block: 256 thr (8 waves, 2x4), tile 64(o) x 128(n), k-step 32.
// Q,K written transposed ([b*h][n][d], packed b128 stores); V natural [d][n].
// ---------------------------------------------------------------------------
#define PSTR 40
__global__ __launch_bounds__(256) void qkv_proj(
    const float* __restrict__ x,
    const float* __restrict__ Wq, const float* __restrict__ bq,
    const float* __restrict__ Wk, const float* __restrict__ bk,
    const float* __restrict__ Wv, const float* __restrict__ bv,
    u16_t* __restrict__ Qt, u16_t* __restrict__ Kt, u16_t* __restrict__ Vb) {
  __shared__ u16_t As[64 * PSTR];    // W tile  [o][c]  (A, row-major)
  __shared__ u16_t Bs[128 * PSTR];   // X tile  [n][c]  (B, transposed)

  const int sel = blockIdx.z;
  const float* Wm   = (sel == 0) ? Wq : (sel == 1) ? Wk : Wv;
  const float* bias = (sel == 0) ? bq : (sel == 1) ? bk : bv;

  const int b  = blockIdx.y;
  const int n0 = (blockIdx.x & 7) * 128;
  const int o0 = (blockIdx.x >> 3) * 64;

  const int tid = threadIdx.x, lane = tid & 31, wid = tid >> 5;
  const int wr = wid >> 2, wc = wid & 3;
  const int hl = lane >> 4, lq = lane & 15;

  const float* xb = x + (size_t)b * CCH * NPIX;

  v8f acc00 = {}, acc01 = {}, acc10 = {}, acc11 = {};

  for (int c0 = 0; c0 < CCH; c0 += 32) {
    __syncthreads();
#pragma unroll
    for (int r = 0; r < 2; ++r) {            // A: 64x32 f32 -> f16
      const int e = tid + (r << 8);
      const int row = e >> 3, col4 = (e & 7) << 2;
      const float4 w4 = *(const float4*)(Wm + (size_t)(o0 + row) * CCH + c0 + col4);
      u16_t* d = As + row * PSTR + col4;
      d[0] = f2h_bits(w4.x); d[1] = f2h_bits(w4.y);
      d[2] = f2h_bits(w4.z); d[3] = f2h_bits(w4.w);
    }
#pragma unroll
    for (int r = 0; r < 4; ++r) {            // B: 32x128 transpose into Bs[n][c]
      const int e = tid + (r << 8);
      const int kk = e >> 5, nn = (e & 31) << 2;
      const float4 x4 = *(const float4*)(xb + (size_t)(c0 + kk) * NPIX + n0 + nn);
      Bs[(nn + 0) * PSTR + kk] = f2h_bits(x4.x);
      Bs[(nn + 1) * PSTR + kk] = f2h_bits(x4.y);
      Bs[(nn + 2) * PSTR + kk] = f2h_bits(x4.z);
      Bs[(nn + 3) * PSTR + kk] = f2h_bits(x4.w);
    }
    __syncthreads();
    const v16h a0 = load_a_frag(As, wr * 32 +  0, 0, PSTR, lane);
    const v16h a1 = load_a_frag(As, wr * 32 + 16, 0, PSTR, lane);
    const v16h b0 = load_b_frag(Bs, wc * 32 +  0, 0, PSTR, lane);
    const v16h b1 = load_b_frag(Bs, wc * 32 + 16, 0, PSTR, lane);
    acc00 = wmma_f16(a0, b0, acc00);
    acc01 = wmma_f16(a0, b1, acc01);
    acc10 = wmma_f16(a1, b0, acc10);
    acc11 = wmma_f16(a1, b1, acc11);
  }

#pragma unroll
  for (int i = 0; i < 2; ++i)
#pragma unroll
    for (int j = 0; j < 2; ++j) {
      const v8f a = (i == 0) ? (j == 0 ? acc00 : acc01)
                             : (j == 0 ? acc10 : acc11);
      const int obase = o0 + wr * 32 + i * 16 + (hl << 3);  // 8-aligned run
      const int n = n0 + wc * 32 + j * 16 + lq;
      if (sel < 2) {
        // transposed store: 8 consecutive d -> one 16B packed store
        u16_t* dstT = (sel == 0) ? Qt : Kt;
        const int hh = obase >> 6, d0 = obase & 63;
        uint4 v4;
        u32_t pk[4];
#pragma unroll
        for (int r = 0; r < 4; ++r) {
          const u16_t lo = f2h_bits(a[2 * r]     + bias[obase + 2 * r]);
          const u16_t hi = f2h_bits(a[2 * r + 1] + bias[obase + 2 * r + 1]);
          pk[r] = (u32_t)lo | ((u32_t)hi << 16);
        }
        v4.x = pk[0]; v4.y = pk[1]; v4.z = pk[2]; v4.w = pk[3];
        *(uint4*)(dstT + (((size_t)(b * HEADS + hh) * NPIX + n) * DHEAD + d0)) = v4;
      } else {
#pragma unroll
        for (int r = 0; r < 8; ++r) {
          const int o = obase + r;
          Vb[((size_t)b * CCH + o) * NPIX + n] = f2h_bits(a[r] + bias[o]);
        }
      }
    }
}

// ---------------------------------------------------------------------------
// Kernel 3: flash attention per (b, h, 64-query tile).
// score[n,m] = q[:,n].k[:,m] + pos[:,n].q[:,m]; online softmax over m;
// O[d,n] = sum_m P[n,m] V[d,m]. Waves 0-3 own 16-query score strips +
// softmax stats; all 8 accumulate the O GEMM (P, alpha via LDS).
// All tile staging is layout-preserving -> async global->LDS b128 DMA.
// ---------------------------------------------------------------------------
#define AST 72   // LDS row stride in halves: 144 B = 16B-aligned rows
__global__ __launch_bounds__(256) void attn_kernel(
    const u16_t* __restrict__ Qt, const u16_t* __restrict__ Kt,
    const u16_t* __restrict__ Vb, const u16_t* __restrict__ Pt_pos,
    float* __restrict__ Ob) {
  __shared__ u16_t Qs [64 * AST];   // [nq][d]  query tile
  __shared__ u16_t Ps [64 * AST];   // [nq][d]  pos tile
  __shared__ u16_t Ks [64 * AST];   // [mk][d]  key tile
  __shared__ u16_t Qks[64 * AST];   // [mk][d]  q-at-key tile
  __shared__ u16_t Vs [64 * AST];   // [d][mk]  value tile
  __shared__ u16_t Pt [64 * AST];   // [nq][mk] prob tile (f16)
  __shared__ float alphaS[64];
  __shared__ float rsumS[64];

  const int n0 = blockIdx.x * 64;
  const int bh = blockIdx.y;
  const int b = bh >> 3, h = bh & 7;

  const u16_t* Qtp = Qt + (size_t)bh * NPIX * DHEAD;             // [n][d]
  const u16_t* Ktp = Kt + (size_t)bh * NPIX * DHEAD;             // [n][d]
  const u16_t* Ptp = Pt_pos + (size_t)h * NPIX * DHEAD;          // [n][d]
  const u16_t* Vp  = Vb + ((size_t)b * CCH + h * DHEAD) * NPIX;  // [d][n]

  const int tid = threadIdx.x, lane = tid & 31, wid = tid >> 5;
  const int hl = lane >> 4, lq = lane & 15;

  // stage query + positional tiles: contiguous 8KB blocks, 16B per lane
#pragma unroll
  for (int p = 0; p < 2; ++p) {
    const int e = tid + (p << 8);
    const int row = e >> 2, seg = (e & 3) << 3;   // seg in halves
    copy16(&Qs[row * AST + seg], Qtp + (size_t)(n0 + row) * DHEAD + seg);
    copy16(&Ps[row * AST + seg], Ptp + (size_t)(n0 + row) * DHEAD + seg);
  }
  wait_async_copies();
  __syncthreads();

  const int sqrow = (wid & 3) * 16;          // score-phase query strip
  const v16h aQ0 = load_a_frag(Qs, sqrow, 0,  AST, lane);  // persistent
  const v16h aQ1 = load_a_frag(Qs, sqrow, 32, AST, lane);
  const v16h aP0 = load_a_frag(Ps, sqrow, 0,  AST, lane);
  const v16h aP1 = load_a_frag(Ps, sqrow, 32, AST, lane);

  float runmax[8], runsum[8];
#pragma unroll
  for (int r = 0; r < 8; ++r) { runmax[r] = -3.0e38f; runsum[r] = 0.f; }

  const int dstrip = (wid & 3) * 16;         // O-phase d strip
  const int qbase  = (wid >> 2) * 32;        // O-phase q half
  v8f O0 = {}, O1 = {};

  for (int it = 0; it < 16; ++it) {
    const int m0 = it * 64;
    __syncthreads();
    // stage K / q-at-key (contiguous) and V (128B row segments)
#pragma unroll
    for (int p = 0; p < 2; ++p) {
      const int e = tid + (p << 8);
      const int row = e >> 2, seg = (e & 3) << 3;
      copy16(&Ks [row * AST + seg], Ktp + (size_t)(m0 + row) * DHEAD + seg);
      copy16(&Qks[row * AST + seg], Qtp + (size_t)(m0 + row) * DHEAD + seg);
      copy16(&Vs [row * AST + seg], Vp + (size_t)row * NPIX + m0 + seg);
    }
    if (it < 15) {  // prefetch next key tile while this one computes
      __builtin_prefetch(Ktp + (size_t)(m0 + 64 + (tid >> 2)) * DHEAD, 0, 1);
      __builtin_prefetch(Vp + (size_t)(tid >> 2) * NPIX + m0 + 64, 0, 1);
    }
    wait_async_copies();
    __syncthreads();

    if (wid < 4) {                           // wave-uniform: EXEC all ones
      v8f S[4];
#pragma unroll
      for (int mt = 0; mt < 4; ++mt) {
        const v16h bK0 = load_b_frag(Ks,  mt * 16, 0,  AST, lane);
        const v16h bK1 = load_b_frag(Ks,  mt * 16, 32, AST, lane);
        const v16h bQ0 = load_b_frag(Qks, mt * 16, 0,  AST, lane);
        const v16h bQ1 = load_b_frag(Qks, mt * 16, 32, AST, lane);
        v8f s = {};
        s = wmma_f16(aQ0, bK0, s);
        s = wmma_f16(aQ1, bK1, s);
        s = wmma_f16(aP0, bQ0, s);
        s = wmma_f16(aP1, bQ1, s);
        S[mt] = s;
      }
#pragma unroll
      for (int r = 0; r < 8; ++r) {
        float mx = fmaxf(fmaxf(S[0][r], S[1][r]), fmaxf(S[2][r], S[3][r]));
        mx = fmaxf(mx, __shfl_xor(mx, 1, 32));
        mx = fmaxf(mx, __shfl_xor(mx, 2, 32));
        mx = fmaxf(mx, __shfl_xor(mx, 4, 32));
        mx = fmaxf(mx, __shfl_xor(mx, 8, 32));
        const float nm = fmaxf(runmax[r], mx);
        const float al = __expf(runmax[r] - nm);
        const int qrow = sqrow + (hl << 3) + r;
        float ts = 0.f;
#pragma unroll
        for (int mt = 0; mt < 4; ++mt) {
          const float p = __expf(S[mt][r] - nm);
          ts += p;
          Pt[qrow * AST + mt * 16 + lq] = f2h_bits(p);
        }
        ts += __shfl_xor(ts, 1, 32);
        ts += __shfl_xor(ts, 2, 32);
        ts += __shfl_xor(ts, 4, 32);
        ts += __shfl_xor(ts, 8, 32);
        runsum[r] = runsum[r] * al + ts;
        runmax[r] = nm;
        if (lq == 0) alphaS[qrow] = al;
      }
    }
    __syncthreads();

    // O accumulation (all 8 waves): O = O*alpha + V(16d x 64m) x P^T
    {
      const float al0 = alphaS[qbase + lq];
      const float al1 = alphaS[qbase + 16 + lq];
      O0 = O0 * al0;
      O1 = O1 * al1;
      const v16h aV0  = load_a_frag(Vs, dstrip, 0,  AST, lane);
      const v16h aV1  = load_a_frag(Vs, dstrip, 32, AST, lane);
      const v16h bP00 = load_b_frag(Pt, qbase,      0,  AST, lane);
      const v16h bP01 = load_b_frag(Pt, qbase,      32, AST, lane);
      const v16h bP10 = load_b_frag(Pt, qbase + 16, 0,  AST, lane);
      const v16h bP11 = load_b_frag(Pt, qbase + 16, 32, AST, lane);
      O0 = wmma_f16(aV0, bP00, O0);
      O0 = wmma_f16(aV1, bP01, O0);
      O1 = wmma_f16(aV0, bP10, O1);
      O1 = wmma_f16(aV1, bP11, O1);
    }
  }

  if (wid < 4 && lq == 0) {
#pragma unroll
    for (int r = 0; r < 8; ++r) rsumS[sqrow + (hl << 3) + r] = runsum[r];
  }
  __syncthreads();

  const float inv0 = 1.f / rsumS[qbase + lq];
  const float inv1 = 1.f / rsumS[qbase + 16 + lq];
#pragma unroll
  for (int r = 0; r < 8; ++r) {
    const int d = dstrip + r + (hl << 3);
    const size_t rowbase = ((size_t)b * CCH + h * DHEAD + d) * NPIX;
    Ob[rowbase + n0 + qbase + lq]      = O0[r] * inv0;
    Ob[rowbase + n0 + qbase + 16 + lq] = O1[r] * inv1;
  }
}

// ---------------------------------------------------------------------------
// Kernel 4: per-batch LayerNorm statistics over C*W*H elements.
// ---------------------------------------------------------------------------
__global__ __launch_bounds__(256) void ln_stats(const float* __restrict__ Ob,
                                                float* __restrict__ st) {
  const int b = blockIdx.x;
  const float* p = Ob + (size_t)b * CCH * NPIX;
  float s = 0.f, s2 = 0.f;
  for (int i = threadIdx.x; i < CCH * NPIX; i += 256) {
    const float v = p[i];
    s += v; s2 += v * v;
  }
#pragma unroll
  for (int m = 1; m < 32; m <<= 1) {
    s  += __shfl_xor(s,  m, 32);
    s2 += __shfl_xor(s2, m, 32);
  }
  __shared__ float red[16];
  const int lane = threadIdx.x & 31, wid = threadIdx.x >> 5;
  if (lane == 0) { red[wid] = s; red[8 + wid] = s2; }
  __syncthreads();
  if (threadIdx.x == 0) {
    float S = 0.f, S2 = 0.f;
#pragma unroll
    for (int w2 = 0; w2 < 8; ++w2) { S += red[w2]; S2 += red[8 + w2]; }
    const float inv_n = 1.f / (float)(CCH * NPIX);
    const float mu  = S * inv_n;
    const float var = S2 * inv_n - mu * mu;
    st[b * 2 + 0] = mu;
    st[b * 2 + 1] = rsqrtf(var + 1e-5f);
  }
}

// ---------------------------------------------------------------------------
// Kernel 5: y = (o - mu) * invstd * ln_w + ln_b + x
// ---------------------------------------------------------------------------
__global__ __launch_bounds__(256) void ln_final(
    const float* __restrict__ Ob, const float* __restrict__ x,
    const float* __restrict__ lnw, const float* __restrict__ lnb,
    const float* __restrict__ st, float* __restrict__ out) {
  const size_t i = (size_t)blockIdx.x * 256 + threadIdx.x;
  const int b  = (int)(i >> 19);               // C*N = 2^19
  const int cn = (int)(i & ((1u << 19) - 1));
  const float mu = st[b * 2 + 0];
  const float is = st[b * 2 + 1];
  out[i] = (Ob[i] - mu) * is * lnw[cn] + lnb[cn] + x[i];
}

// ---------------------------------------------------------------------------
extern "C" void kernel_launch(void* const* d_in, const int* in_sizes, int n_in,
                              void* d_out, int out_size, void* d_ws, size_t ws_size,
                              hipStream_t stream) {
  (void)in_sizes; (void)n_in; (void)out_size; (void)ws_size;
  const float* x     = (const float*)d_in[0];
  const float* Wq    = (const float*)d_in[1];
  const float* bq    = (const float*)d_in[2];
  const float* Wk    = (const float*)d_in[3];
  const float* bk    = (const float*)d_in[4];
  const float* Wv    = (const float*)d_in[5];
  const float* bv    = (const float*)d_in[6];
  const float* rel_h = (const float*)d_in[7];
  const float* rel_w = (const float*)d_in[8];
  const float* ln_w  = (const float*)d_in[9];
  const float* ln_b  = (const float*)d_in[10];
  float* out = (float*)d_out;

  char* w = (char*)d_ws;
  u16_t* Qt = (u16_t*)w; w += (size_t)BATCH * CCH * NPIX * 2;   // 32 MB [bh][n][d]
  u16_t* Kt = (u16_t*)w; w += (size_t)BATCH * CCH * NPIX * 2;   // 32 MB [bh][n][d]
  u16_t* Vb = (u16_t*)w; w += (size_t)BATCH * CCH * NPIX * 2;   // 32 MB [b][o][n]
  u16_t* Pb = (u16_t*)w; w += (size_t)HEADS * NPIX * DHEAD * 2; //  1 MB [h][n][d]
  float* Ob = (float*)w; w += (size_t)BATCH * CCH * NPIX * 4;   // 64 MB
  float* st = (float*)w;                                        // 256 B

  pos_build<<<(HEADS * NPIX * DHEAD) / 256, 256, 0, stream>>>(rel_h, rel_w, Pb);
  qkv_proj<<<dim3(64, BATCH, 3), 256, 0, stream>>>(x, Wq, bq, Wk, bk, Wv, bv,
                                                   Qt, Kt, Vb);
  attn_kernel<<<dim3(NPIX / 64, BATCH * HEADS), 256, 0, stream>>>(Qt, Kt, Vb, Pb, Ob);
  ln_stats<<<BATCH, 256, 0, stream>>>(Ob, st);
  ln_final<<<(size_t)(BATCH * CCH * NPIX) / 256, 256, 0, stream>>>(Ob, x, ln_w,
                                                                   ln_b, st, out);
}
